// KroneckerProductFusionClassifier_29583734735150
// MI455X (gfx1250) — matrix operand order repaired and checked by reading, not compiled, not run on hardware
//
#include <hip/hip_runtime.h>
#include <hip/hip_bf16.h>
#include <cstdint>

typedef __attribute__((ext_vector_type(8)))  float  v8f;
typedef __attribute__((ext_vector_type(4)))  float  v4f;
typedef __attribute__((ext_vector_type(16))) __bf16 v16bf;
typedef __attribute__((ext_vector_type(4)))  unsigned u32x4;
typedef __attribute__((ext_vector_type(4)))  int      i32x4;
typedef __attribute__((ext_vector_type(8)))  int      i32x8;

#define BATCH  256
#define P_DIMc 512
#define M_DIMc 768
#define XCOLS  1280
#define PROJc  512
#define KTOT   (P_DIMc * M_DIMc)   /* 393216 */
#define NCHUNK (KTOT / 32)         /* 12288 K-chunks of 32 */
#define KSPLIT 32
#define CPW    (NCHUNK / KSPLIT)   /* 384 chunks per workgroup */

__global__ __launch_bounds__(256) void zero_fused_kernel(float* __restrict__ f) {
  int i = blockIdx.x * blockDim.x + threadIdx.x;
  if (i < BATCH * PROJc) f[i] = 0.0f;
}

// ---- Tensor Data Mover: stage a 64-row x 32-float fp32 W1 block into LDS. ----
// 2-D descriptor per CDNA5 ISA ch. 8: tile_dim0=32 elems (contiguous K),
// tile_dim1=64 rows, row stride = KTOT elems, data_size=4B, LDS dest packed
// [row][32] (8 KB). One wave issues it; completion via TENSORcnt.
__device__ __forceinline__ void tdm_stage_w1(const float* __restrict__ g,
                                             unsigned lds_off) {
  unsigned long long ga = (unsigned long long)(uintptr_t)g;
  u32x4 g0;
  g0[0] = 1u;                                    // count=1, user descriptor
  g0[1] = lds_off;                               // lds_addr (bytes)
  g0[2] = (unsigned)ga;                          // global_addr[31:0]
  g0[3] = (unsigned)((ga >> 32) & 0x01ffffffu)   // global_addr[56:32]
          | (2u << 30);                          // type = 2 ("image")
  i32x8 g1;
  g1[0] = 2 << 16;                               // data_size = 4 bytes
  g1[1] = (int)((KTOT & 0xffffu) << 16);         // tensor_dim0[15:0]  (bits 63:48)
  g1[2] = (int)(((unsigned)KTOT >> 16)           // tensor_dim0[31:16] (bits 79:64)
          | (512u << 16));                       // tensor_dim1[15:0]  (bits 95:80)
  g1[3] = (int)(32u << 16);                      // tensor_dim1[31:16]=0 | tile_dim0=32
  g1[4] = 64;                                    // tile_dim1=64, tile_dim2=0
  g1[5] = (int)KTOT;                             // tensor_dim0_stride[31:0]
  g1[6] = 0;                                     // stride[47:32] | dim1_stride lo = 0
  g1[7] = 0;
  i32x4 z4 = {0, 0, 0, 0};
#if defined(__clang_major__) && (__clang_major__ >= 23)
  i32x8 z8 = {0, 0, 0, 0, 0, 0, 0, 0};
  __builtin_amdgcn_tensor_load_to_lds(g0, g1, z4, z4, z8, 0);
#else
  __builtin_amdgcn_tensor_load_to_lds(g0, g1, z4, z4, 0);
#endif
}

// Build a 16x32 bf16 A-fragment (one batch tile) for v_wmma_f32_16x16x32_bf16.
// A layout (16-bit, wave32): lanes 0-15 -> M=lane, K halves {0..7} in v0-3 and
// {16..23} in v4-7; lanes 16-31 -> M=lane-16, K halves {8..15} and {24..31}.
// m is pre-offset by (mm0 + half*8): m[0..7] first K-half group, m[16..23] second.
__device__ __forceinline__ v16bf make_a_frag(float pv, const float* __restrict__ m) {
  v4f m0 = *(const v4f*)(m);
  v4f m1 = *(const v4f*)(m + 4);
  v4f m2 = *(const v4f*)(m + 16);
  v4f m3 = *(const v4f*)(m + 20);
  v16bf a;
#pragma unroll
  for (int i = 0; i < 4; ++i) {
    a[i]      = (__bf16)(pv * m0[i]);
    a[4 + i]  = (__bf16)(pv * m1[i]);
    a[8 + i]  = (__bf16)(pv * m2[i]);
    a[12 + i] = (__bf16)(pv * m3[i]);
  }
  return a;
}

// Phase 1: fused(256x512) += Z(256xK) * W1^T, Z formed on the fly from x.
// Grid: (8 n-groups of 64 cols, KSPLIT k-splits). Block: 256 threads = 8 waves.
// Wave w: batch rows [32w, 32w+32) as two 16-row M tiles, 4 N tiles of 16.
// W1 is TDM-staged to LDS (double buffered), so each 8 KB block hits VMEM once
// per workgroup instead of once per wave.
__global__ __launch_bounds__(256) void kron_gemm1_kernel(
    const float* __restrict__ x, const float* __restrict__ W1,
    float* __restrict__ fused) {
  __shared__ float sW[2][64 * 32];          // 2 x 8 KB W1 staging buffers

  const int lane = threadIdx.x & 31;
  const int wave = threadIdx.x >> 5;
  const int lm   = lane & 15;
  const int half = lane >> 4;

  const int kbase = blockIdx.x * 64;        // first of 64 output columns
  const int b0a   = wave * 32;              // first batch tile
  const int b0b   = b0a + 16;               // second batch tile

  v8f acc[4][2] = {};                       // [ktile][batch tile]

  const float* __restrict__ xa = x + (size_t)(b0a + lm) * XCOLS;
  const float* __restrict__ xb = x + (size_t)(b0b + lm) * XCOLS;
  const int mbase = half * 8;               // A-fragment m offset

  const float* __restrict__ wbase = W1 + (size_t)kbase * KTOT;
  const int chunk0 = blockIdx.y * CPW;

  if (threadIdx.x < 32)
    tdm_stage_w1(wbase + (size_t)chunk0 * 32, (unsigned)(uintptr_t)&sW[0][0]);

  for (int c = 0; c < CPW; ++c) {
    const int K   = (chunk0 + c) * 32;      // 768 % 32 == 0 -> chunk has one pp
    const int buf = c & 1;

    if (threadIdx.x < 32) __builtin_amdgcn_s_wait_tensorcnt(0);
    __syncthreads();                        // staged data visible to all waves;
                                            // also: everyone done reading buf^1
    if ((c + 1 < CPW) && (threadIdx.x < 32))
      tdm_stage_w1(wbase + (size_t)K + 32,  // DMA next chunk while we compute
                   (unsigned)(uintptr_t)&sW[buf ^ 1][0]);

    const int pp  = K / M_DIMc;
    const int mm0 = K - pp * M_DIMc;
    const float pa = xa[pp];
    const float pb = xb[pp];
    v16bf A0 = make_a_frag(pa, xa + P_DIMc + mm0 + mbase);
    v16bf A1 = make_a_frag(pb, xb + P_DIMc + mm0 + mbase);

#pragma unroll
    for (int j = 0; j < 4; ++j) {
      // B layout (16-bit, 32x16): lanes 0-15 -> N=lane, K {0..7} and {16..23};
      // lanes 16-31 -> N=lane-16, K {8..15} and {24..31} (matches A).
      const int srow = (j * 16 + lm) * 32 + half * 8;
      v4f w0 = *(const v4f*)&sW[buf][srow];
      v4f w1 = *(const v4f*)&sW[buf][srow + 4];
      v4f w2 = *(const v4f*)&sW[buf][srow + 16];
      v4f w3 = *(const v4f*)&sW[buf][srow + 20];
      v16bf B;
#pragma unroll
      for (int i = 0; i < 4; ++i) {
        B[i]      = (__bf16)w0[i];
        B[4 + i]  = (__bf16)w1[i];
        B[8 + i]  = (__bf16)w2[i];
        B[12 + i] = (__bf16)w3[i];
      }
      acc[j][0] = __builtin_amdgcn_wmma_f32_16x16x32_bf16(
          false, A0, false, B, (short)0, acc[j][0], false, false);
      acc[j][1] = __builtin_amdgcn_wmma_f32_16x16x32_bf16(
          false, A1, false, B, (short)0, acc[j][1], false, false);
    }
  }

  // C/D layout: VGPR r -> lanes 0-15: (M=r, N=lane); lanes 16-31: (M=r+8, N=lane-16)
  const int col = kbase + lm;
#pragma unroll
  for (int j = 0; j < 4; ++j) {
#pragma unroll
    for (int r = 0; r < 8; ++r) {
      const int rowa = b0a + r + half * 8;
      const int rowb = b0b + r + half * 8;
      unsafeAtomicAdd(&fused[(size_t)rowa * PROJc + col + j * 16], acc[j][0][r]);
      unsafeAtomicAdd(&fused[(size_t)rowb * PROJc + col + j * 16], acc[j][1][r]);
    }
  }
}

// Phase 2: per-row LayerNorm + relu, then 512->256->128->2 dense tail.
__global__ __launch_bounds__(256) void tail_kernel(
    const float* __restrict__ fused, const float* __restrict__ b1,
    const float* __restrict__ ln_g, const float* __restrict__ ln_b,
    const float* __restrict__ W2, const float* __restrict__ b2,
    const float* __restrict__ bn1_g, const float* __restrict__ bn1_b,
    const float* __restrict__ W3, const float* __restrict__ b3,
    const float* __restrict__ bn2_g, const float* __restrict__ bn2_b,
    const float* __restrict__ W4, const float* __restrict__ b4,
    float* __restrict__ out) {
  __shared__ float h1[512];
  __shared__ float h2[256];
  __shared__ float h3[128];
  __shared__ float red[8];
  const int row = blockIdx.x;
  const int t   = threadIdx.x;
  const float rs = rsqrtf(1.0f + 1e-5f);   // bn-eval scale 1/sqrt(1+eps)

  float v0 = fused[(size_t)row * 512 + t] + b1[t];
  float v1 = fused[(size_t)row * 512 + 256 + t] + b1[256 + t];

  float s = v0 + v1;
#pragma unroll
  for (int o = 16; o > 0; o >>= 1) s += __shfl_xor(s, o, 32);
  if ((t & 31) == 0) red[t >> 5] = s;
  __syncthreads();
  float tot = 0.0f;
#pragma unroll
  for (int i = 0; i < 8; ++i) tot += red[i];
  const float mu = tot * (1.0f / 512.0f);
  __syncthreads();

  const float d0 = v0 - mu, d1 = v1 - mu;
  s = d0 * d0 + d1 * d1;
#pragma unroll
  for (int o = 16; o > 0; o >>= 1) s += __shfl_xor(s, o, 32);
  if ((t & 31) == 0) red[t >> 5] = s;
  __syncthreads();
  tot = 0.0f;
#pragma unroll
  for (int i = 0; i < 8; ++i) tot += red[i];
  const float inv = rsqrtf(tot * (1.0f / 512.0f) + 1e-5f);

  h1[t]       = fmaxf(d0 * inv * ln_g[t] + ln_b[t], 0.0f);
  h1[256 + t] = fmaxf(d1 * inv * ln_g[256 + t] + ln_b[256 + t], 0.0f);
  __syncthreads();

  {
    float a = 0.0f;
    const float* __restrict__ w = W2 + (size_t)t * 512;
    for (int i = 0; i < 512; ++i) a = fmaf(h1[i], w[i], a);
    a += b2[t];
    h2[t] = fmaxf(a * rs * bn1_g[t] + bn1_b[t], 0.0f);
  }
  __syncthreads();

  if (t < 128) {
    float a = 0.0f;
    const float* __restrict__ w = W3 + (size_t)t * 256;
    for (int i = 0; i < 256; ++i) a = fmaf(h2[i], w[i], a);
    a += b3[t];
    h3[t] = fmaxf(a * rs * bn2_g[t] + bn2_b[t], 0.0f);
  }
  __syncthreads();

  if (t < 2) {
    float a = 0.0f;
    const float* __restrict__ w = W4 + (size_t)t * 128;
    for (int i = 0; i < 128; ++i) a = fmaf(h3[i], w[i], a);
    out[(size_t)row * 2 + t] = a + b4[t];
  }
}

extern "C" void kernel_launch(void* const* d_in, const int* in_sizes, int n_in,
                              void* d_out, int out_size, void* d_ws, size_t ws_size,
                              hipStream_t stream) {
  (void)in_sizes; (void)n_in; (void)out_size; (void)ws_size;
  const float* x     = (const float*)d_in[0];
  const float* W1    = (const float*)d_in[1];
  const float* b1    = (const float*)d_in[2];
  const float* ln_g  = (const float*)d_in[3];
  const float* ln_b  = (const float*)d_in[4];
  const float* W2    = (const float*)d_in[5];
  const float* b2    = (const float*)d_in[6];
  const float* bn1_g = (const float*)d_in[7];
  const float* bn1_b = (const float*)d_in[8];
  const float* W3    = (const float*)d_in[9];
  const float* b3    = (const float*)d_in[10];
  const float* bn2_g = (const float*)d_in[11];
  const float* bn2_b = (const float*)d_in[12];
  const float* W4    = (const float*)d_in[13];
  const float* b4    = (const float*)d_in[14];
  float* out   = (float*)d_out;
  float* fused = (float*)d_ws;             // 256*512 floats = 512 KB scratch

  zero_fused_kernel<<<(BATCH * PROJc + 255) / 256, 256, 0, stream>>>(fused);
  kron_gemm1_kernel<<<dim3(PROJc / 64, KSPLIT), 256, 0, stream>>>(x, W1, fused);
  tail_kernel<<<BATCH, 256, 0, stream>>>(fused, b1, ln_g, ln_b, W2, b2, bn1_g,
                                         bn1_b, W3, b3, bn2_g, bn2_b, W4, b4, out);
}
